// BERTVideo_DividedSpaceTimeAttn_68934225101527
// MI455X (gfx1250) — matrix operand order repaired and checked by reading, not compiled, not run on hardware
//
#include <hip/hip_runtime.h>
#include <hip/hip_bf16.h>

// Problem constants (from reference): E=256, H=8, HD=32, N = 1 + 64*1024 = 65537
#define E      256
#define H      8
#define NROWS  65537
#define NPAD   65664            // 513 * 128, row-tile padded
#define RSQRT_HD 0.17677669529663687f   // 1/sqrt(32)

typedef __attribute__((ext_vector_type(16))) __bf16 v16bf;
typedef __attribute__((ext_vector_type(8)))  __bf16 v8bf;
typedef __attribute__((ext_vector_type(8)))  float  v8f;
typedef __attribute__((ext_vector_type(4)))  unsigned int u32x4;
typedef __attribute__((ext_vector_type(8)))  int i32x8;
typedef __attribute__((ext_vector_type(4)))  int i32x4;

#if defined(__gfx1250__) && __has_builtin(__builtin_amdgcn_tensor_load_to_lds) && \
    __has_builtin(__builtin_amdgcn_s_wait_tensorcnt)
#define USE_TDM 1
#else
#define USE_TDM 0
#endif

__device__ __forceinline__ float waveRedSum(float v) {
#pragma unroll
    for (int o = 16; o > 0; o >>= 1) v += __shfl_xor(v, o, 32);
    return v;
}
__device__ __forceinline__ float waveRedMax(float v) {
#pragma unroll
    for (int o = 16; o > 0; o >>= 1) v = fmaxf(v, __shfl_xor(v, o, 32));
    return v;
}

// ---- scalar sums of the 6 projection weight tensors (each H*32*32 = 8192 floats)
__global__ __launch_bounds__(256)
void k_wsum(const float* w0, const float* w1, const float* w2,
            const float* w3, const float* w4, const float* w5, float* sw) {
    const float* p;
    switch (blockIdx.x) {
        case 0: p = w0; break; case 1: p = w1; break; case 2: p = w2; break;
        case 3: p = w3; break; case 4: p = w4; break; default: p = w5; break;
    }
    float s = 0.f;
    for (int i = threadIdx.x; i < H * 32 * 32; i += 256) s += p[i];
    s = waveRedSum(s);
    __shared__ float red[8];
    if ((threadIdx.x & 31) == 0) red[threadIdx.x >> 5] = s;
    __syncthreads();
    if (threadIdx.x == 0) {
        float t = 0.f;
        for (int i = 0; i < 8; i++) t += red[i];
        sw[blockIdx.x] = t;
    }
}

// ---- layernorm: one row per block (256 threads). Wave w == head w (32 dims).
__global__ __launch_bounds__(256)
void k_ln(const float* __restrict__ x, const float* __restrict__ g,
          const float* __restrict__ bta, __bf16* __restrict__ ybf,
          float* __restrict__ nrm) {
    const size_t row = blockIdx.x;
    const int tid = threadIdx.x, lane = tid & 31, wv = tid >> 5;
    float v = x[row * E + tid];
    __shared__ float red[8], red2[8];
    float s = waveRedSum(v);
    if (!lane) red[wv] = s;
    __syncthreads();
    float tot = red[0] + red[1] + red[2] + red[3] + red[4] + red[5] + red[6] + red[7];
    float mean = tot * (1.f / E);
    float d = v - mean;
    float ss = waveRedSum(d * d);
    if (!lane) red2[wv] = ss;
    __syncthreads();
    float var = (red2[0] + red2[1] + red2[2] + red2[3] +
                 red2[4] + red2[5] + red2[6] + red2[7]) * (1.f / E);
    float y = d * rsqrtf(var + 1e-5f) * g[tid] + bta[tid];
    ybf[row * E + tid] = (__bf16)y;
    float q = waveRedSum(y * y);
    if (!lane) nrm[row * H + wv] = q;
}

// ---- softmax stats per (a, h): block = a, wave = head. Scores over {token, b=0..d1-1}.
__global__ __launch_bounds__(256)
void k_softmax(const float* __restrict__ nrm, const float* __restrict__ sw,
               int iq, int ik, float* __restrict__ mOut, float* __restrict__ rzOut,
               float* __restrict__ aw0Out, int d1) {
    const int a = blockIdx.x, h = threadIdx.x >> 5, lane = threadIdx.x & 31;
    const float qk = sw[iq] * sw[ik] * RSQRT_HD;
    const float st = qk * nrm[h];   // token score (row 0)
    const float* np = nrm + ((size_t)a * d1 + 1) * H + h;
    float mx = st;
    for (int b = lane; b < d1; b += 32) mx = fmaxf(mx, qk * np[(size_t)b * H]);
    mx = waveRedMax(mx);
    float z = (lane == 0) ? __expf(st - mx) : 0.f;
    for (int b = lane; b < d1; b += 32) z += __expf(qk * np[(size_t)b * H] - mx);
    z = waveRedSum(z);
    if (!lane) {
        mOut[a * H + h] = mx;
        rzOut[a * H + h] = 1.f / z;
        aw0Out[a * H + h] = __expf(st - mx) / z;
    }
}

__global__ void k_zero(float* p) { p[threadIdx.x] = 0.f; }

// ---- build bf16 attention-output rows 1..N-1; accumulate token-row reduction.
__global__ __launch_bounds__(256)
void k_build(const __bf16* __restrict__ ybf, const float* __restrict__ nrm,
             const float* __restrict__ sw, int iq, int ik, int iv,
             const float* __restrict__ mB, const float* __restrict__ rzB,
             const float* __restrict__ aw0B, __bf16* __restrict__ outbf,
             float* __restrict__ tok, int log2d1) {
    const int e = threadIdx.x, h = e >> 5;
    const float qk = sw[iq] * sw[ik] * RSQRT_HD;
    const float sv = sw[iv];
    const float y0 = (float)ybf[e];
    const size_t gbase = (size_t)blockIdx.x * 64;
    float acc = 0.f;
    for (int i = 0; i < 64; i++) {
        size_t gidx = gbase + i, row = gidx + 1;
        int a = (int)(gidx >> log2d1);
        float yv = (float)ybf[row * E + e];
        float aw1 = __expf(qk * nrm[row * H + h] - mB[a * H + h]) * rzB[a * H + h];
        float a0 = aw0B[a * H + h];
        outbf[row * E + e] = (__bf16)(sv * (aw1 * yv + a0 * y0));
        acc += a0 * yv;
    }
    atomicAdd(&tok[e], acc);
}

// ---- finalize token row: out[0] = sv*(y0 + sum aw0*y)
__global__ void k_fin(const __bf16* __restrict__ ybf, const float* __restrict__ tok,
                      const float* __restrict__ sw, int iv, __bf16* __restrict__ outbf) {
    int e = threadIdx.x;
    outbf[e] = (__bf16)(sw[iv] * ((float)ybf[e] + tok[e]));
}

// ---- convert 256x256 fp32 weight into bf16 B-fragment-swizzled layout:
// dst[((ct*8+c)*32 + L)*16 + i]  holds  B[K = c*32 + (i/8)*16 + (L/16)*8 + (i&7)][N = ct*16 + (L&15)]
__global__ __launch_bounds__(256)
void k_prepw(const float* __restrict__ W, __bf16* __restrict__ wswz, int transpose) {
    int idx = blockIdx.x * 256 + threadIdx.x;
    int i = idx & 15, L = (idx >> 4) & 31, c = (idx >> 9) & 7, ct = idx >> 12;
    int K = c * 32 + (i >> 3) * 16 + (L >> 4) * 8 + (i & 7);
    int Nc = ct * 16 + (L & 15);
    float v = transpose ? W[Nc * E + K] : W[K * E + Nc];
    wswz[idx] = (__bf16)v;
}

// ---- GEMM: out(N,256) = A_bf16(N,256) @ B(256,256) + resid + bias
// Whole swizzled B staged in 128KB dynamic LDS via the Tensor Data Mover once per
// block; B fragments stream from LDS pipelined against the 8-deep WMMA chain.
// bias is always a valid pointer (zero buffer when the reference has no bias),
// and the full/partial row-tile split is hoisted out of the column-tile loop.
__global__ __launch_bounds__(256)
void k_gemm(const __bf16* __restrict__ A, const __bf16* __restrict__ Bs,
            const float* __restrict__ resid, const float* __restrict__ bias,
            float* __restrict__ out, int nRows) {
    extern __shared__ __bf16 shB[];          // E*E bf16 = 128 KB dynamic LDS
    const int tid = threadIdx.x, lane = tid & 31, wv = tid >> 5;

#if USE_TDM
    if (tid < 32) {                           // wave 0 issues the TDM transfer
        unsigned long long ga = (unsigned long long)(const void*)Bs;
        unsigned int laddr = (unsigned int)(unsigned long long)(const void*)shB;
        // D# group 0: count=1, lds_addr, global_addr[56:0], type=2 ("image")
        u32x4 g0 = { 1u, laddr, (unsigned int)(ga & 0xFFFFFFFFu),
                     (unsigned int)((ga >> 32) & 0x01FFFFFFu) | 0x80000000u };
        // D# group 1: data_size=1 (2B), 2D tensor 256x256, tile 256x256, stride0=256
        i32x8 g1 = { (int)(1u << 16),           // wg_mask=0, data_size=1
                     (int)(256u << 16),         // tensor_dim0[15:0]=256 in bits[63:48]
                     (int)(256u << 16),         // tensor_dim0 hi=0, tensor_dim1[15:0]=256
                     (int)(256u << 16),         // tensor_dim1 hi=0, tile_dim0=256
                     256,                       // tile_dim1=256, tile_dim2=0
                     256,                       // tensor_dim0_stride[31:0]=256
                     0, 0 };
        i32x4 gz4 = { 0, 0, 0, 0 };
        i32x8 gz8 = { 0, 0, 0, 0, 0, 0, 0, 0 };
        __builtin_amdgcn_tensor_load_to_lds(g0, g1, gz4, gz4, gz8, 0);
        __builtin_amdgcn_s_wait_tensorcnt(0);
    }
#else
    {   // cooperative fallback copy (also compiled by the host pass)
        const v8bf* src = (const v8bf*)Bs;
        v8bf* dst = (v8bf*)shB;
        for (int i = tid; i < E * E / 8; i += 256) dst[i] = src[i];
    }
#endif
    __syncthreads();

    const int r0 = blockIdx.x * 128 + wv * 16;
    const int mrow = lane & 15, hi = lane >> 4;
    int rowA = r0 + mrow;
    rowA = rowA < nRows ? rowA : (nRows - 1);   // clamp (stores are guarded)
    const __bf16* aptr = A + (size_t)rowA * E + hi * 8;

    // CDNA5 speculative prefetch of a future A tile (global_prefetch_b8)
    __builtin_prefetch((const void*)(A + (size_t)(r0 + 1024) * E), 0, 0);

    // A fragment per K-chunk: lane L holds M=L%16; elems 0..7 -> K = k0 + 8*(L/16) + 0..7,
    // elems 8..15 -> K = k0 + 16 + 8*(L/16) + 0..7  (two contiguous b128 loads).
    v16bf afrag[8];
#pragma unroll
    for (int c = 0; c < 8; c++) {
        v8bf lo = *(const v8bf*)(aptr + c * 32);
        v8bf hb = *(const v8bf*)(aptr + c * 32 + 16);
        afrag[c] = __builtin_shufflevector(lo, hb, 0, 1, 2, 3, 4, 5, 6, 7,
                                           8, 9, 10, 11, 12, 13, 14, 15);
    }

    const int col = lane & 15, mh = hi * 8;
    if ((r0 + 16) <= nRows) {
        // ---- hot path: full 16-row tile, unguarded stores
        for (int ct = 0; ct < 16; ct++) {
            v16bf bfrag[8];
            const __bf16* bptr = shB + ((size_t)ct * 8 * 32 + lane) * 16;
#pragma unroll
            for (int c = 0; c < 8; c++)
                bfrag[c] = *(const v16bf*)(bptr + (size_t)c * 32 * 16);

            v8f acc = {0.f, 0.f, 0.f, 0.f, 0.f, 0.f, 0.f, 0.f};
#pragma unroll
            for (int c = 0; c < 8; c++)
                acc = __builtin_amdgcn_wmma_f32_16x16x32_bf16(
                    false, afrag[c], false, bfrag[c], (short)0, acc, false, false);

            const int nc = ct * 16 + col;
            const float bb = bias[nc];
#pragma unroll
            for (int r = 0; r < 8; r++) {
                int rr = r0 + mh + r;   // C/D layout: VGPR r -> M = r + 8*(lane/16)
                out[(size_t)rr * E + nc] = acc[r] + resid[(size_t)rr * E + nc] + bb;
            }
        }
    } else {
        // ---- tail path: last row tile, per-row guards
        for (int ct = 0; ct < 16; ct++) {
            v16bf bfrag[8];
            const __bf16* bptr = shB + ((size_t)ct * 8 * 32 + lane) * 16;
#pragma unroll
            for (int c = 0; c < 8; c++)
                bfrag[c] = *(const v16bf*)(bptr + (size_t)c * 32 * 16);

            v8f acc = {0.f, 0.f, 0.f, 0.f, 0.f, 0.f, 0.f, 0.f};
#pragma unroll
            for (int c = 0; c < 8; c++)
                acc = __builtin_amdgcn_wmma_f32_16x16x32_bf16(
                    false, afrag[c], false, bfrag[c], (short)0, acc, false, false);

            const int nc = ct * 16 + col;
            const float bb = bias[nc];
#pragma unroll
            for (int r = 0; r < 8; r++) {
                int rr = r0 + mh + r;
                if (rr < nRows)
                    out[(size_t)rr * E + nc] = acc[r] + resid[(size_t)rr * E + nc] + bb;
            }
        }
    }
}

extern "C" void kernel_launch(void* const* d_in, const int* in_sizes, int n_in,
                              void* d_out, int out_size, void* d_ws, size_t ws_size,
                              hipStream_t stream) {
    (void)in_sizes; (void)n_in; (void)out_size; (void)ws_size;
    const float* emb   = (const float*)d_in[0];
    const float* lntg  = (const float*)d_in[1];
    const float* lntb  = (const float*)d_in[2];
    const float* Wq_t  = (const float*)d_in[3];
    const float* Wk_t  = (const float*)d_in[4];
    const float* Wv_t  = (const float*)d_in[5];
    const float* Wt_t  = (const float*)d_in[6];
    const float* lnsg  = (const float*)d_in[7];
    const float* lnsb  = (const float*)d_in[8];
    const float* Wq_s  = (const float*)d_in[9];
    const float* Wk_s  = (const float*)d_in[10];
    const float* Wv_s  = (const float*)d_in[11];
    const float* Wt_s  = (const float*)d_in[12];
    const float* lnmg  = (const float*)d_in[13];
    const float* lnmb  = (const float*)d_in[14];
    const float* W_mlp = (const float*)d_in[15];
    const float* b_mlp = (const float*)d_in[16];

    char* ws = (char*)d_ws;
    size_t off = 0;
    auto nxt = [&](size_t bytes) -> char* {
        char* p = ws + off;
        off += (bytes + 255) & ~(size_t)255;
        return p;
    };
    float*  sw    = (float*)nxt(6 * sizeof(float));
    float*  tok   = (float*)nxt(E * sizeof(float));
    float*  bzero = (float*)nxt(E * sizeof(float));
    float*  mB    = (float*)nxt((size_t)1024 * H * sizeof(float));
    float*  rzB   = (float*)nxt((size_t)1024 * H * sizeof(float));
    float*  aw0B  = (float*)nxt((size_t)1024 * H * sizeof(float));
    float*  nrm   = (float*)nxt((size_t)NROWS * H * sizeof(float));
    __bf16* ybf   = (__bf16*)nxt((size_t)NPAD * E * sizeof(__bf16));
    __bf16* obf   = (__bf16*)nxt((size_t)NPAD * E * sizeof(__bf16));
    __bf16* wswz  = (__bf16*)nxt((size_t)E * E * sizeof(__bf16));
    float*  p1    = (float*)nxt((size_t)NROWS * E * sizeof(float));
    float*  p2    = (float*)nxt((size_t)NROWS * E * sizeof(float));

    const size_t ldsB = (size_t)E * E * sizeof(__bf16);   // 128 KB dynamic LDS

    k_wsum<<<6, 256, 0, stream>>>(Wq_t, Wk_t, Wv_t, Wq_s, Wk_s, Wv_s, sw);
    k_zero<<<1, 256, 0, stream>>>(bzero);   // zero bias for phases 1-2

    // Phase 1: temporal attention (d0 = P = 1024, d1 = B = 64)
    k_ln<<<NROWS, 256, 0, stream>>>(emb, lntg, lntb, ybf, nrm);
    k_softmax<<<1024, 256, 0, stream>>>(nrm, sw, 0, 1, mB, rzB, aw0B, 64);
    k_zero<<<1, 256, 0, stream>>>(tok);
    k_build<<<1024, 256, 0, stream>>>(ybf, nrm, sw, 0, 1, 2, mB, rzB, aw0B, obf, tok, 6);
    k_fin<<<1, 256, 0, stream>>>(ybf, tok, sw, 2, obf);
    k_prepw<<<256, 256, 0, stream>>>(Wt_t, wswz, 0);
    k_gemm<<<NPAD / 128, 256, ldsB, stream>>>(obf, wswz, emb, bzero, p1, NROWS);

    // Phase 2: spatial attention (d0 = B = 64, d1 = P = 1024)
    k_ln<<<NROWS, 256, 0, stream>>>(p1, lnsg, lnsb, ybf, nrm);
    k_softmax<<<64, 256, 0, stream>>>(nrm, sw, 3, 4, mB, rzB, aw0B, 1024);
    k_zero<<<1, 256, 0, stream>>>(tok);
    k_build<<<1024, 256, 0, stream>>>(ybf, nrm, sw, 3, 4, 5, mB, rzB, aw0B, obf, tok, 10);
    k_fin<<<1, 256, 0, stream>>>(ybf, tok, sw, 5, obf);
    k_prepw<<<256, 256, 0, stream>>>(Wt_s, wswz, 0);
    k_gemm<<<NPAD / 128, 256, ldsB, stream>>>(obf, wswz, p1, bzero, p2, NROWS);

    // Phase 3: final LN + MLP (W transposed) + bias + residual
    k_ln<<<NROWS, 256, 0, stream>>>(p2, lnmg, lnmb, ybf, nrm);
    k_prepw<<<256, 256, 0, stream>>>(W_mlp, wswz, 1);
    k_gemm<<<NPAD / 128, 256, ldsB, stream>>>(ybf, wswz, p2, b_mlp, (float*)d_out, NROWS);
}
